// LlamaAttention_51230369907419
// MI455X (gfx1250) — matrix operand order repaired and checked by reading, not compiled, not run on hardware
//
#include <hip/hip_runtime.h>
#include <stdint.h>

// ---------------------------------------------------------------------------
// Types for CDNA5 WMMA (wave32): D(f32 16x16) = A(bf16 16x32) x B(bf16 32x16) + C
// ---------------------------------------------------------------------------
typedef __attribute__((ext_vector_type(16))) __bf16          v16bf;
typedef __attribute__((ext_vector_type(8)))  float           v8f;
typedef __attribute__((ext_vector_type(8)))  unsigned short  v8u16;
typedef __attribute__((ext_vector_type(16))) unsigned short  v16u16;

#define SEQ    2048
#define HIDDEN 2048
#define HQ     32
#define HKV    4
#define DHEAD  64
#define NQKV   (HQ + 2 * HKV)            // 40 heads in qkv projection
#define QKV_N  (NQKV * DHEAD)            // 2560

// ---- f32 -> bf16 (round-to-nearest-even) via bit trick
__device__ __forceinline__ unsigned short f2bf(float x) {
  unsigned int u = __float_as_uint(x);
  u += 0x7FFFu + ((u >> 16) & 1u);
  return (unsigned short)(u >> 16);
}

__device__ __forceinline__ v16bf pack_frag(v8u16 lo, v8u16 hi) {
  v16u16 u = __builtin_shufflevector(lo, hi, 0, 1, 2, 3, 4, 5, 6, 7,
                                     8, 9, 10, 11, 12, 13, 14, 15);
  union { v16u16 u; v16bf b; } f;
  f.u = u;
  return f.b;
}

__device__ __forceinline__ v16bf cast_frag(v16u16 u) {
  union { v16u16 u; v16bf b; } f;
  f.u = u;
  return f.b;
}

// ---- LDS byte-offset of a __shared__ pointer (generic -> AS(3) -> u32)
typedef __attribute__((address_space(3))) const unsigned char lds_cu8;
__device__ __forceinline__ unsigned lds_off_of(const void* p) {
  return (unsigned)(unsigned long long)(lds_cu8*)p;
}

// ---- CDNA5 async global->LDS copy, tracked with ASYNCcnt
__device__ __forceinline__ void async_copy_b128(const unsigned short* gsrc,
                                                const unsigned short* ldst) {
  unsigned loff = lds_off_of(ldst);
  unsigned long long ga = (unsigned long long)gsrc;
  asm volatile("global_load_async_to_lds_b128 %0, %1, off"
               :: "v"(loff), "v"(ga) : "memory");
}

__device__ __forceinline__ void wait_async0() {
  asm volatile("s_wait_asynccnt 0x0" ::: "memory");
}

// ---------------------------------------------------------------------------
// Kernel 1: f32 -> bf16 conversion (4 elems / thread)
// ---------------------------------------------------------------------------
__global__ void cvt_f32_bf16(const float* __restrict__ in,
                             unsigned short* __restrict__ out, int n) {
  int i = (blockIdx.x * blockDim.x + threadIdx.x) * 4;
  if (i + 3 < n) {
    float4 v = *(const float4*)(in + i);
    out[i + 0] = f2bf(v.x);
    out[i + 1] = f2bf(v.y);
    out[i + 2] = f2bf(v.z);
    out[i + 3] = f2bf(v.w);
  }
}

// ---------------------------------------------------------------------------
// Kernel 2/5: NT GEMM, C[M,N] f32 = A[M,K] bf16 (row-major) * B[N,K] bf16.
// Block = 4 waves (2M x 2N) -> 128x64 block tile; wave tile 64x32
// (4x2 accumulators = 64 VGPRs -> no spills). The block's B tile (64 rows x
// 32 K bf16 = 4 KB) is staged into LDS via async global->LDS copies,
// double-buffered: issue next step's copies, compute 8 WMMAs on current
// buffer, then s_wait_asynccnt 0 (own-wave) + barrier (cross-wave publish).
// ---------------------------------------------------------------------------
__global__ void __launch_bounds__(128)
gemm_nt_wmma(const unsigned short* __restrict__ A,
             const unsigned short* __restrict__ B,
             float* __restrict__ C,
             int M, int N, int K) {
  const int lane = threadIdx.x & 31;
  const int wave = threadIdx.x >> 5;
  const int col  = lane & 15;   // lane within 16-group
  const int half = lane >> 4;   // 0: lanes 0-15, 1: lanes 16-31
  const int wm   = wave >> 1;   // wave row (M)
  const int wn   = wave & 1;    // wave col (N)
  const int m0   = blockIdx.y * 128 + wm * 64;
  const int nblk = blockIdx.x * 64;

  __shared__ __align__(16) unsigned short Bsm[2][64 * 32];

  // stage one 64x32 B tile (4 KB) into Bsm[buf]: 256 x 16B chunks, 2/thread
  auto stage = [&](int buf, int kb) {
    int t = threadIdx.x;
#pragma unroll
    for (int u = 0; u < 2; ++u) {
      int c   = t + u * 128;   // 0..255
      int r   = c >> 2;        // B-tile row (n-local, 0..63)
      int seg = c & 3;         // 16B segment within 32-elem row
      const unsigned short* g = B + (size_t)(nblk + r) * K + kb + seg * 8;
      async_copy_b128(g, &Bsm[buf][r * 32 + seg * 8]);
    }
  };

  v8f zero = {};
  v8f acc[4][2];
#pragma unroll
  for (int i = 0; i < 4; ++i)
#pragma unroll
    for (int j = 0; j < 2; ++j) acc[i][j] = zero;

  const unsigned short* arow[4];
#pragma unroll
  for (int i = 0; i < 4; ++i) arow[i] = A + (size_t)(m0 + 16 * i + col) * K;

  stage(0, 0);
  wait_async0();
  __syncthreads();

  int cur = 0;
  for (int kb = 0; kb < K; kb += 32) {
    if (kb + 32 < K) stage(cur ^ 1, kb + 32);

    v16bf af[4], bfr[2];
#pragma unroll
    for (int i = 0; i < 4; ++i) {
      // A 16x32 bf16 frag: lane row=col, K = half*16 + 0..15 contiguous
      const unsigned short* p = arow[i] + kb + half * 16;
      af[i] = pack_frag(*(const v8u16*)p, *(const v8u16*)(p + 8));
    }
#pragma unroll
    for (int j = 0; j < 2; ++j) {
      // B 32x16 bf16 frag from LDS: lane col, K = half*16 + 0..15 contiguous
      const unsigned short* p = &Bsm[cur][(wn * 32 + j * 16 + col) * 32 + half * 16];
      bfr[j] = cast_frag(*(const v16u16*)p);
    }
#pragma unroll
    for (int i = 0; i < 4; ++i)
#pragma unroll
      for (int j = 0; j < 2; ++j)
        acc[i][j] = __builtin_amdgcn_wmma_f32_16x16x32_bf16(
            false, af[i], false, bfr[j], (short)0, acc[i][j], false, false);

    wait_async0();     // own-wave async copies landed in LDS
    __syncthreads();   // publish to the other waves / protect WAR on Bsm
    cur ^= 1;
  }

#pragma unroll
  for (int i = 0; i < 4; ++i)
#pragma unroll
    for (int j = 0; j < 2; ++j) {
      int cc = nblk + wn * 32 + 16 * j + col;
#pragma unroll
      for (int t = 0; t < 8; ++t) {
        int rr = m0 + 16 * i + t + 8 * half;  // C tile: vgpr t -> row t (+8 hi lanes)
        C[(size_t)rr * N + cc] = acc[i][j][t];
      }
    }
}

// ---------------------------------------------------------------------------
// Kernel 3: RoPE + scatter to attention-friendly bf16 layouts.
//   Q -> [HQ][S][D] row-major, K -> [HKV][S][D], V -> transposed [HKV][D][S].
// ---------------------------------------------------------------------------
__global__ void rope_scatter(const float* __restrict__ qkv,  // [S][QKV_N]
                             unsigned short* __restrict__ Qb,
                             unsigned short* __restrict__ Kb,
                             unsigned short* __restrict__ Vt) {
  int idx = blockIdx.x * blockDim.x + threadIdx.x;  // S * 40 * 32
  int i = idx & 31;
  int h = (idx >> 5) % NQKV;
  int s = idx / (NQKV * 32);
  if (s >= SEQ) return;

  const float* base = qkv + (size_t)s * QKV_N + h * DHEAD;
  float x1 = base[i];
  float x2 = base[i + 32];

  if (h < HQ + HKV) {
    float freq = __powf(10000.0f, -(float)i * (1.0f / 32.0f));
    float ang  = (float)s * freq;
    float sn, cs;
    __sincosf(ang, &sn, &cs);
    float r1 = x1 * cs - x2 * sn;
    float r2 = x2 * cs + x1 * sn;
    if (h < HQ) {
      unsigned short* q = Qb + ((size_t)h * SEQ + s) * DHEAD;
      q[i] = f2bf(r1);
      q[i + 32] = f2bf(r2);
    } else {
      unsigned short* k = Kb + ((size_t)(h - HQ) * SEQ + s) * DHEAD;
      k[i] = f2bf(r1);
      k[i + 32] = f2bf(r2);
    }
  } else {
    unsigned short* v = Vt + (size_t)(h - HQ - HKV) * DHEAD * SEQ;
    v[(size_t)i * SEQ + s] = f2bf(x1);
    v[(size_t)(i + 32) * SEQ + s] = f2bf(x2);
  }
}

// ---------------------------------------------------------------------------
// Kernel 4: causal flash attention.
// Grid (HQ, SEQ/64); block = 128 (4 waves). Each wave: 16 query rows x D=64.
// Per 64-key block: S=Q*K^T (8 wmmas), online softmax in f32 registers,
// P restaged through LDS (C-tile -> A-fragment re-layout), O+=P*V (8 wmmas).
// Key-loop trip count is uniform per workgroup -> __syncthreads is legal.
// ---------------------------------------------------------------------------
__global__ void __launch_bounds__(128)
attn_flash(const unsigned short* __restrict__ Q,   // [HQ][S][D]
           const unsigned short* __restrict__ Km,  // [HKV][S][D]
           const unsigned short* __restrict__ Vt,  // [HKV][D][S]
           unsigned short* __restrict__ Out) {     // [S][HQ*D]
  const int hq   = blockIdx.x;
  const int kvh  = hq >> 3;  // HQ/HKV = 8
  const int qblk = blockIdx.y;
  const int wave = threadIdx.x >> 5;
  const int lane = threadIdx.x & 31;
  const int col  = lane & 15;
  const int half = lane >> 4;
  const int q0   = qblk * 64 + wave * 16;
  const float scale = 0.125f;  // 1/sqrt(64)

  __shared__ __align__(32) unsigned short Pbuf[4][16 * 64];

  // Q A-fragments (2 chunks of K=32 over D=64), row = q0+col.
  v16bf qf[2];
  {
    const unsigned short* qrow = Q + ((size_t)hq * SEQ + q0 + col) * DHEAD;
#pragma unroll
    for (int kc = 0; kc < 2; ++kc) {
      const unsigned short* p = qrow + kc * 32 + half * 16;
      qf[kc] = pack_frag(*(const v8u16*)p, *(const v8u16*)(p + 8));
    }
  }

  float rm[8], rs[8];
  v8f zero = {};
  v8f o_acc[4];
#pragma unroll
  for (int t = 0; t < 8; ++t) { rm[t] = -3.0e38f; rs[t] = 0.0f; }
#pragma unroll
  for (int j = 0; j < 4; ++j) o_acc[j] = zero;

  const int nblocks = qblk + 1;  // causal: keys up to q0_block+63
  for (int b = 0; b < nblocks; ++b) {
    const int kb = b * 64;

    // ---- S = Q*K^T for 64 keys (4 N-tiles x 2 K-chunks)
    v8f sacc[4];
#pragma unroll
    for (int j = 0; j < 4; ++j) sacc[j] = zero;
#pragma unroll
    for (int j = 0; j < 4; ++j) {
      const unsigned short* kr =
          Km + ((size_t)kvh * SEQ + kb + j * 16 + col) * DHEAD;
#pragma unroll
      for (int kc = 0; kc < 2; ++kc) {
        v16bf kfrag = cast_frag(*(const v16u16*)(kr + kc * 32 + half * 16));
        sacc[j] = __builtin_amdgcn_wmma_f32_16x16x32_bf16(
            false, qf[kc], false, kfrag, (short)0, sacc[j], false, false);
      }
    }

    // ---- scale + causal mask + block row-max
    float bm[8];
#pragma unroll
    for (int t = 0; t < 8; ++t) bm[t] = rm[t];
#pragma unroll
    for (int j = 0; j < 4; ++j) {
#pragma unroll
      for (int t = 0; t < 8; ++t) {
        float v = sacc[j][t] * scale;
        int key = kb + j * 16 + col;
        int qi  = q0 + t + 8 * half;
        if (key > qi) v = -3.0e38f;
        sacc[j][t] = v;
        bm[t] = fmaxf(bm[t], v);
      }
    }
#pragma unroll
    for (int m = 1; m <= 8; m <<= 1)
#pragma unroll
      for (int t = 0; t < 8; ++t)
        bm[t] = fmaxf(bm[t], __shfl_xor(bm[t], m, 32));

    // ---- online softmax update
    float alpha[8], ps[8];
#pragma unroll
    for (int t = 0; t < 8; ++t) {
      alpha[t] = __expf(rm[t] - bm[t]);
      rm[t] = bm[t];
      ps[t] = 0.0f;
    }
#pragma unroll
    for (int j = 0; j < 4; ++j) {
#pragma unroll
      for (int t = 0; t < 8; ++t) {
        float p = __expf(sacc[j][t] - bm[t]);
        sacc[j][t] = p;
        ps[t] += p;
      }
    }
#pragma unroll
    for (int m = 1; m <= 8; m <<= 1)
#pragma unroll
      for (int t = 0; t < 8; ++t) ps[t] += __shfl_xor(ps[t], m, 32);
#pragma unroll
    for (int t = 0; t < 8; ++t) rs[t] = rs[t] * alpha[t] + ps[t];
#pragma unroll
    for (int j = 0; j < 4; ++j)
#pragma unroll
      for (int t = 0; t < 8; ++t) o_acc[j][t] *= alpha[t];

    // ---- P: C-tile layout -> LDS row-major [16][64] bf16
#pragma unroll
    for (int j = 0; j < 4; ++j)
#pragma unroll
      for (int t = 0; t < 8; ++t)
        Pbuf[wave][(t + 8 * half) * 64 + j * 16 + col] = f2bf(sacc[j][t]);
    __syncthreads();

    // ---- P A-fragments from LDS
    v16bf pf[2];
    {
      const unsigned short* prow = &Pbuf[wave][(size_t)col * 64];
#pragma unroll
      for (int kc = 0; kc < 2; ++kc) {
        const unsigned short* p = prow + kc * 32 + half * 16;
        pf[kc] = pack_frag(*(const v8u16*)p, *(const v8u16*)(p + 8));
      }
    }

    // ---- O += P * V  (V^T layout gives contiguous K per d-column)
#pragma unroll
    for (int j = 0; j < 4; ++j) {
      const unsigned short* vcol =
          Vt + ((size_t)kvh * DHEAD + j * 16 + col) * SEQ + kb;
#pragma unroll
      for (int kc = 0; kc < 2; ++kc) {
        v16bf vfrag = cast_frag(*(const v16u16*)(vcol + kc * 32 + half * 16));
        o_acc[j] = __builtin_amdgcn_wmma_f32_16x16x32_bf16(
            false, pf[kc], false, vfrag, (short)0, o_acc[j], false, false);
      }
    }
    __syncthreads();  // WAR on Pbuf before next block
  }

  // ---- epilogue: O /= l, write bf16 [s][hq*64+d]
#pragma unroll
  for (int j = 0; j < 4; ++j) {
#pragma unroll
    for (int t = 0; t < 8; ++t) {
      float o = o_acc[j][t] / rs[t];
      int row = q0 + t + 8 * half;
      int c   = hq * DHEAD + j * 16 + col;
      Out[(size_t)row * (HQ * DHEAD) + c] = f2bf(o);
    }
  }
}

// ---------------------------------------------------------------------------
// Host-side launcher. Workspace layout (bytes, 64 MB total).
// ---------------------------------------------------------------------------
extern "C" void kernel_launch(void* const* d_in, const int* in_sizes, int n_in,
                              void* d_out, int out_size, void* d_ws, size_t ws_size,
                              hipStream_t stream) {
  const float* hs   = (const float*)d_in[0];  // [1][2048][2048]
  const float* wqkv = (const float*)d_in[1];  // [2560][2048]
  const float* wo   = (const float*)d_in[2];  // [2048][2048]
  float* out = (float*)d_out;                 // [1][2048][2048] f32

  uint8_t* ws = (uint8_t*)d_ws;
  unsigned short* hsb   = (unsigned short*)(ws + 0);           //  8 MB  hidden bf16
  unsigned short* wqkvb = (unsigned short*)(ws + 8388608);     // 10 MB  w_qkv bf16
  unsigned short* wob   = (unsigned short*)(ws + 18874368);    //  8 MB  w_o bf16
  float*          qkvf  = (float*)(ws + 27262976);             // 20 MB  qkv f32
  unsigned short* qb    = (unsigned short*)(ws + 48234496);    //  8 MB  Q roped bf16
  unsigned short* kbuf  = (unsigned short*)(ws + 56623104);    //  1 MB  K roped bf16
  unsigned short* vtb   = (unsigned short*)(ws + 57671680);    //  1 MB  V^T bf16
  unsigned short* aob   = (unsigned short*)(ws + 58720256);    //  8 MB  attn out bf16

  // 1) f32 -> bf16 conversions
  cvt_f32_bf16<<<(HIDDEN * SEQ) / 1024, 256, 0, stream>>>(hs, hsb, HIDDEN * SEQ);
  cvt_f32_bf16<<<(QKV_N * HIDDEN) / 1024, 256, 0, stream>>>(wqkv, wqkvb, QKV_N * HIDDEN);
  cvt_f32_bf16<<<(HIDDEN * HIDDEN) / 1024, 256, 0, stream>>>(wo, wob, HIDDEN * HIDDEN);

  // 2) QKV projection: [2048,2048] x [2560,2048]^T -> [2048,2560] f32
  gemm_nt_wmma<<<dim3(QKV_N / 64, SEQ / 128), 128, 0, stream>>>(
      hsb, wqkvb, qkvf, SEQ, QKV_N, HIDDEN);

  // 3) RoPE + scatter to Q/K/V^T bf16 layouts
  rope_scatter<<<(SEQ * NQKV * 32) / 256, 256, 0, stream>>>(qkvf, qb, kbuf, vtb);

  // 4) causal flash attention
  attn_flash<<<dim3(HQ, SEQ / 64), 128, 0, stream>>>(qb, kbuf, vtb, aob);

  // 5) output projection: [2048,2048] x [2048,2048]^T -> d_out f32
  gemm_nt_wmma<<<dim3(HIDDEN / 64, SEQ / 128), 128, 0, stream>>>(
      aob, wob, out, SEQ, HIDDEN, HIDDEN);
}
